// HyperSAGPool_60601988547129
// MI455X (gfx1250) — matrix operand order repaired and checked by reading, not compiled, not run on hardware
//
#include <hip/hip_runtime.h>
#include <math.h>

// ---------------- problem constants (from reference) ----------------
#define NUM_GRAPHS 8
#define N_PER      65536
#define NUM_NODES  (NUM_GRAPHS * N_PER)   // 524288
#define C_IN       128
#define NNZ        4194304
#define NUM_EDGES  262144
#define KSEL       52429                  // ceil(0.8 * 65536)
#define NKEEP      (NUM_GRAPHS * KSEL)    // 419432

typedef float v2f  __attribute__((ext_vector_type(2)));
typedef float v8f  __attribute__((ext_vector_type(8)));
typedef _Float16 v16h __attribute__((ext_vector_type(16)));
typedef unsigned long long u64;

// ---------------- fills ----------------
__global__ void k_fill_f32(float* p, float v, int n) {
    int i = blockIdx.x * blockDim.x + threadIdx.x;
    if (i < n) p[i] = v;
}
__global__ void k_fill_i32(int* p, int v, int n) {
    int i = blockIdx.x * blockDim.x + threadIdx.x;
    if (i < n) p[i] = v;
}

// ---------------- GEMV xl = x @ W via f32 WMMA ----------------
// One wave32 -> 16 rows. A = 16x4 f32 tile of x; B = 4x16 with W in col 0.
// Column-0 masking is done arithmetically (wsel) so there is NO control flow:
// EXEC stays all-ones (WMMA requirement) and loads pipeline freely.
// After 32 accumulations (K=128), D column 0 (lanes 0 and 16) = row dots.
#if __has_builtin(__builtin_amdgcn_wmma_f32_16x16x4_f32)
__global__ void k_gemv(const float* __restrict__ x, const float* __restrict__ W,
                       float* __restrict__ xl) {
    int gtid = blockIdx.x * blockDim.x + threadIdx.x;
    int wave = gtid >> 5;
    int lane = threadIdx.x & 31;
    int half = lane >> 4;       // 0: K pair {0,1}, 1: K pair {2,3}
    int r    = lane & 15;       // A: row M; B/D: column N
    long rowBase = (long)wave * 16;
    const float* xrow = x + (rowBase + r) * C_IN + 2 * half;
    const float* wrow = W + 2 * half;
    float wsel = (r == 0) ? 1.0f : 0.0f;   // B[K][N] = W[K] iff N==0
    v8f c = {};
    #pragma unroll
    for (int k0 = 0; k0 < C_IN; k0 += 4) {
        v2f a, bm;
        a.x  = xrow[k0];
        a.y  = xrow[k0 + 1];
        bm.x = wrow[k0] * wsel;
        bm.y = wrow[k0 + 1] * wsel;
        c = __builtin_amdgcn_wmma_f32_16x16x4_f32(false, a, false, bm,
                                                  (short)0, c, false, false);
    }
    if (r == 0) {               // lanes 0 (rows 0-7) and 16 (rows 8-15) hold col 0
        float* dst = xl + rowBase + 8 * half;
        #pragma unroll
        for (int v = 0; v < 8; ++v) dst[v] = c[v];
    }
}
#else
// fallback: codegen-confirmed f16 WMMA (16x16x32), 4 K-steps
__global__ void k_gemv(const float* __restrict__ x, const float* __restrict__ W,
                       float* __restrict__ xl) {
    int gtid = blockIdx.x * blockDim.x + threadIdx.x;
    int wave = gtid >> 5;
    int lane = threadIdx.x & 31;
    int half = lane >> 4;
    int r    = lane & 15;
    long rowBase = (long)wave * 16;
    const float* xrow = x + (rowBase + r) * C_IN;
    float wsel = (r == 0) ? 1.0f : 0.0f;
    v8f c = {};
    for (int k0 = 0; k0 < C_IN; k0 += 32) {
        v16h a, bm;
        #pragma unroll
        for (int e = 0; e < 16; ++e) {
            int v = e >> 1, pbit = e & 1;
            int kA = (v < 4 ? 2 * v : 16 + 2 * (v - 4)) + 8 * half + pbit;
            a[e]  = (_Float16)xrow[k0 + kA];
            int kB = (e < 8 ? e : e + 8) + 8 * half;
            bm[e] = (_Float16)(W[k0 + kB] * wsel);
        }
        c = __builtin_amdgcn_wmma_f32_16x16x32_f16(false, a, false, bm,
                                                   (short)0, c, false, false);
    }
    if (r == 0) {
        float* dst = xl + rowBase + 8 * half;
        #pragma unroll
        for (int v = 0; v < 8; ++v) dst[v] = c[v];
    }
}
#endif

// ---------------- incidence accumulation ----------------
__global__ void k_edge_accum(const int* __restrict__ nodes, const int* __restrict__ edges,
                             const float* __restrict__ xl,
                             float* edge_deg, float* node_deg, float* edge_sum) {
    int j = blockIdx.x * blockDim.x + threadIdx.x;
    if (j >= NNZ) return;
    int n = nodes[j], e = edges[j];
    atomicAdd(edge_deg + e, 1.0f);
    atomicAdd(node_deg + n, 1.0f);
    atomicAdd(edge_sum + e, xl[n]);
}

__global__ void k_edge_norm(float* edge_sum, const float* __restrict__ edge_deg) {
    int e = blockIdx.x * blockDim.x + threadIdx.x;
    if (e >= NUM_EDGES) return;
    float d = edge_deg[e];
    edge_sum[e] = (d > 0.0f) ? edge_sum[e] / d : 0.0f;
}

__global__ void k_node_accum(const int* __restrict__ nodes, const int* __restrict__ edges,
                             const float* __restrict__ edge_feat, float* node_sum) {
    int j = blockIdx.x * blockDim.x + threadIdx.x;
    if (j >= NNZ) return;
    atomicAdd(node_sum + nodes[j], edge_feat[edges[j]]);
}

// score = Dinv * node_sum + b ; build 64-bit sort key (descending score, ties by idx)
__global__ void k_score_keys(const float* __restrict__ node_deg,
                             const float* __restrict__ node_sum,
                             const float* __restrict__ b,
                             float* __restrict__ score, u64* __restrict__ keys) {
    int i = blockIdx.x * blockDim.x + threadIdx.x;
    if (i >= NUM_NODES) return;
    float nd = node_deg[i];
    float s = ((nd > 0.0f) ? node_sum[i] / nd : 0.0f) + b[0];
    score[i] = s;
    unsigned u = __float_as_uint(s);
    unsigned m = (u & 0x80000000u) ? ~u : (u | 0x80000000u); // monotone ascending map
    unsigned dkey = ~m;                                      // descending
    keys[i] = ((u64)dkey << 32) | (unsigned)(i & (N_PER - 1));
}

// ---------------- per-graph bitonic sort (segments of 65536) ----------------
__global__ void k_bitonic(u64* keys, int j, int k) {
    int gid = blockIdx.x * blockDim.x + threadIdx.x;   // [0, NUM_NODES)
    int seg = gid >> 16;
    int i   = gid & 0xFFFF;
    int l   = i ^ j;
    if (l > i) {
        u64* base = keys + ((size_t)seg << 16);
        u64 a = base[i], b = base[l];
        bool up = ((i & k) == 0);
        if ((a > b) == up) { base[i] = b; base[l] = a; }
    }
}

// ---------------- perm / batch_new / node_mask ----------------
__global__ void k_perm(const u64* __restrict__ keys, const int* __restrict__ batch,
                       int* __restrict__ perm_int, int* __restrict__ node_mask,
                       float* __restrict__ out_perm, float* __restrict__ out_batch) {
    int gi = blockIdx.x * blockDim.x + threadIdx.x;
    if (gi >= NKEEP) return;
    int g = gi / KSEL, i = gi % KSEL;
    int local = (int)(unsigned)(keys[(size_t)g * N_PER + i] & 0xFFFFFFFFull);
    int p = g * N_PER + local;
    perm_int[gi]  = p;
    node_mask[p]  = gi;
    out_perm[gi]  = (float)p;
    out_batch[gi] = (float)batch[p];
}

// x_new[r] = x[perm[r]] * tanh(score[perm[r]]) — float4 vectorized
__global__ void k_xnew(const float* __restrict__ x, const float* __restrict__ score,
                       const int* __restrict__ perm_int, float* __restrict__ out) {
    int tid = blockIdx.x * blockDim.x + threadIdx.x;   // NKEEP*32
    if (tid >= NKEEP * 32) return;
    int row = tid >> 5, q = tid & 31;
    int p = perm_int[row];
    float t = tanhf(score[p]);
    const float4* xs = (const float4*)x;
    float4 v = xs[(size_t)p * 32 + q];
    v.x *= t; v.y *= t; v.z *= t; v.w *= t;
    ((float4*)out)[(size_t)row * 32 + q] = v;
}

// ---------------- incidence filtering ----------------
__global__ void k_keep(const int* __restrict__ nodes, const int* __restrict__ edges,
                       const int* __restrict__ node_mask,
                       int* __restrict__ keepPos, int* __restrict__ edge_cnt) {
    int j = blockIdx.x * blockDim.x + threadIdx.x;
    if (j >= NNZ) return;
    int kq = (node_mask[nodes[j]] >= 0) ? 1 : 0;
    keepPos[j] = kq;
    if (kq) atomicAdd(edge_cnt + edges[j], 1);
}

__global__ void k_edge_flag(const int* __restrict__ edge_cnt, int* __restrict__ edge_cum) {
    int e = blockIdx.x * blockDim.x + threadIdx.x;
    if (e >= NUM_EDGES) return;
    edge_cum[e] = (edge_cnt[e] > 0) ? 1 : 0;
}

// inclusive block scan (Hillis-Steele in LDS), block size 512
#define SCAN_B 512
__global__ void k_scan_block(const int* __restrict__ in, int* __restrict__ out,
                             int* bsums, int n) {
    __shared__ int sh[SCAN_B];
    int t = threadIdx.x;
    int i = blockIdx.x * SCAN_B + t;
    sh[t] = (i < n) ? in[i] : 0;
    __syncthreads();
    #pragma unroll
    for (int off = 1; off < SCAN_B; off <<= 1) {
        int tv = (t >= off) ? sh[t - off] : 0;
        __syncthreads();
        sh[t] += tv;
        __syncthreads();
    }
    if (i < n) out[i] = sh[t];
    if (t == SCAN_B - 1 && bsums) bsums[blockIdx.x] = sh[SCAN_B - 1];
}

__global__ void k_scan_add(int* out, const int* __restrict__ bsums_scanned, int n) {
    int i = blockIdx.x * SCAN_B + threadIdx.x;
    if (blockIdx.x > 0 && i < n) out[i] += bsums_scanned[blockIdx.x - 1];
}

__global__ void k_compact(const int* __restrict__ nodes, const int* __restrict__ edges,
                          const int* __restrict__ node_mask, const int* __restrict__ keepPos,
                          const int* __restrict__ edge_cum, float* __restrict__ hidx) {
    int j = blockIdx.x * blockDim.x + threadIdx.x;
    if (j >= NNZ) return;
    int n2 = node_mask[nodes[j]];
    if (n2 >= 0) {
        int pos = keepPos[j] - 1;                 // inclusive -> exclusive
        int e2  = edge_cum[edges[j]] - 1;         // renumbered edge id
        hidx[pos]       = (float)n2;
        hidx[NNZ + pos] = (float)e2;
    }
}

// ---------------- launch ----------------
extern "C" void kernel_launch(void* const* d_in, const int* in_sizes, int n_in,
                              void* d_out, int out_size, void* d_ws, size_t ws_size,
                              hipStream_t stream) {
    (void)in_sizes; (void)n_in; (void)out_size; (void)ws_size;
    const float* x     = (const float*)d_in[0];
    const int*   hinc  = (const int*)d_in[1];
    const int*   nodes = hinc;
    const int*   edges = hinc + NNZ;
    const int*   batch = (const int*)d_in[2];
    const float* W     = (const float*)d_in[3];
    const float* b     = (const float*)d_in[4];
    float* outf = (float*)d_out;

    // d_out layout (floats): x_new | hidx(2*NNZ) | batch_new | perm
    const size_t HIDX_OFF  = (size_t)NKEEP * C_IN;        // 53,687,296
    const size_t BATCH_OFF = HIDX_OFF + (size_t)2 * NNZ;  // 62,075,904
    const size_t PERM_OFF  = BATCH_OFF + NKEEP;           // 62,495,336

    // d_ws layout (256B aligned)
    char* ws = (char*)d_ws;
    size_t off = 0;
    auto alloc = [&](size_t bytes) { char* p = ws + off; off = (off + bytes + 255) & ~(size_t)255; return p; };
    float* score     = (float*)alloc((size_t)NUM_NODES * 4);
    float* node_deg  = (float*)alloc((size_t)NUM_NODES * 4);
    float* node_sum  = (float*)alloc((size_t)NUM_NODES * 4);
    float* edge_deg  = (float*)alloc((size_t)NUM_EDGES * 4);
    float* edge_sum  = (float*)alloc((size_t)NUM_EDGES * 4);
    u64*   keys      = (u64*)  alloc((size_t)NUM_NODES * 8);
    int*   node_mask = (int*)  alloc((size_t)NUM_NODES * 4);
    int*   perm_int  = (int*)  alloc((size_t)NKEEP * 4);
    int*   keepPos   = (int*)  alloc((size_t)NNZ * 4);
    int*   edge_cnt  = (int*)  alloc((size_t)NUM_EDGES * 4);
    int*   edge_cum  = (int*)  alloc((size_t)NUM_EDGES * 4);
    int*   bsum0     = (int*)  alloc(8192 * 4);
    int*   bsum1     = (int*)  alloc(512 * 4);
    int*   bsumE     = (int*)  alloc(512 * 4);

    const int B = 256;

    // init
    k_fill_f32<<<(NUM_NODES + B - 1) / B, B, 0, stream>>>(node_deg, 0.0f, NUM_NODES);
    k_fill_f32<<<(NUM_NODES + B - 1) / B, B, 0, stream>>>(node_sum, 0.0f, NUM_NODES);
    k_fill_f32<<<(NUM_EDGES + B - 1) / B, B, 0, stream>>>(edge_deg, 0.0f, NUM_EDGES);
    k_fill_f32<<<(NUM_EDGES + B - 1) / B, B, 0, stream>>>(edge_sum, 0.0f, NUM_EDGES);
    k_fill_i32<<<(NUM_EDGES + B - 1) / B, B, 0, stream>>>(edge_cnt, 0, NUM_EDGES);
    k_fill_i32<<<(NUM_NODES + B - 1) / B, B, 0, stream>>>(node_mask, -1, NUM_NODES);
    k_fill_f32<<<(2 * NNZ + B - 1) / B, B, 0, stream>>>(outf + HIDX_OFF, -1.0f, 2 * NNZ);

    // 1) xl = x @ W  (WMMA; one wave -> 16 rows; 8 waves/block -> 128 rows)
    k_gemv<<<NUM_NODES / 128, 256, 0, stream>>>(x, W, score);

    // 2) degrees + edge aggregation, normalize, node aggregation, score + keys
    k_edge_accum<<<NNZ / B, B, 0, stream>>>(nodes, edges, score, edge_deg, node_deg, edge_sum);
    k_edge_norm<<<NUM_EDGES / B, B, 0, stream>>>(edge_sum, edge_deg);
    k_node_accum<<<NNZ / B, B, 0, stream>>>(nodes, edges, edge_sum, node_sum);
    k_score_keys<<<NUM_NODES / B, B, 0, stream>>>(node_deg, node_sum, b, score, keys);

    // 3) per-graph bitonic sort (descending score, ties by index)
    for (int k = 2; k <= N_PER; k <<= 1)
        for (int j = k >> 1; j >= 1; j >>= 1)
            k_bitonic<<<NUM_NODES / B, B, 0, stream>>>(keys, j, k);

    // 4) perm / batch_new / node remap, gated gather
    k_perm<<<(NKEEP + B - 1) / B, B, 0, stream>>>(keys, batch, perm_int, node_mask,
                                                  outf + PERM_OFF, outf + BATCH_OFF);
    k_xnew<<<(NKEEP * 32 + B - 1) / B, B, 0, stream>>>(x, score, perm_int, outf);

    // 5) incidence filter: keep flags + edge counts
    k_keep<<<NNZ / B, B, 0, stream>>>(nodes, edges, node_mask, keepPos, edge_cnt);
    k_edge_flag<<<NUM_EDGES / B, B, 0, stream>>>(edge_cnt, edge_cum);

    // edge renumber scan: 262144 -> 512 blocks -> single block
    k_scan_block<<<NUM_EDGES / SCAN_B, SCAN_B, 0, stream>>>(edge_cum, edge_cum, bsumE, NUM_EDGES);
    k_scan_block<<<1, SCAN_B, 0, stream>>>(bsumE, bsumE, nullptr, NUM_EDGES / SCAN_B);
    k_scan_add<<<NUM_EDGES / SCAN_B, SCAN_B, 0, stream>>>(edge_cum, bsumE, NUM_EDGES);

    // keep-position scan: 4194304 -> 8192 -> 16 -> 1
    k_scan_block<<<NNZ / SCAN_B, SCAN_B, 0, stream>>>(keepPos, keepPos, bsum0, NNZ);
    k_scan_block<<<8192 / SCAN_B, SCAN_B, 0, stream>>>(bsum0, bsum0, bsum1, 8192);
    k_scan_block<<<1, SCAN_B, 0, stream>>>(bsum1, bsum1, nullptr, 8192 / SCAN_B);
    k_scan_add<<<8192 / SCAN_B, SCAN_B, 0, stream>>>(bsum0, bsum1, 8192);
    k_scan_add<<<NNZ / SCAN_B, SCAN_B, 0, stream>>>(keepPos, bsum0, NNZ);

    // 6) stable compaction into hidx (tail already -1)
    k_compact<<<NNZ / B, B, 0, stream>>>(nodes, edges, node_mask, keepPos, edge_cum,
                                         outf + HIDX_OFF);
}
